// OrderModule_20942260535550
// MI455X (gfx1250) — matrix operand order repaired and checked by reading, not compiled, not run on hardware
//
#include <hip/hip_runtime.h>
#include <hip/hip_bf16.h>

typedef __attribute__((ext_vector_type(8)))  __bf16 v8bf;
typedef __attribute__((ext_vector_type(16))) __bf16 v16bf;
typedef __attribute__((ext_vector_type(8)))  float  v8f;

static __host__ __device__ inline int cdiv(long long a, long long b) { return (int)((a + b - 1) / b); }

// ---------------------------------------------------------------------------
// Utility elementwise kernels
// ---------------------------------------------------------------------------
__global__ void k_fill(float* p, float v, int n) {
  int i = blockIdx.x * blockDim.x + threadIdx.x;
  if (i < n) p[i] = v;
}

__global__ void k_f32_to_bf16(const float* __restrict__ in, __bf16* __restrict__ out, int n) {
  int i = blockIdx.x * blockDim.x + threadIdx.x;
  if (i < n) out[i] = (__bf16)in[i];
}

// in: [K,N] row-major fp32  ->  out: [N,K] row-major bf16
__global__ void k_transpose_f32_to_bf16(const float* __restrict__ in, __bf16* __restrict__ out,
                                        int K, int N) {
  int i = blockIdx.x * blockDim.x + threadIdx.x;
  if (i >= K * N) return;
  int k = i / N, n = i - k * N;
  out[(long long)n * K + k] = (__bf16)in[i];
}

// v: [B*R, D] bf16 -> vT: [B, D, R] bf16   (per-head columns become contiguous rows)
__global__ void k_transpose_v_heads(const __bf16* __restrict__ v, __bf16* __restrict__ vT,
                                    int B_, int R_, int D_) {
  int i = blockIdx.x * blockDim.x + threadIdx.x;
  int total = B_ * R_ * D_;
  if (i >= total) return;
  int b = i / (R_ * D_);
  int rem = i - b * R_ * D_;
  int r = rem / D_;
  int d = rem - r * D_;
  vT[(long long)b * D_ * R_ + (long long)d * R_ + r] = v[i];
}

__global__ void k_gather_emb_bf16(const int* __restrict__ idx, const float* __restrict__ emb,
                                  __bf16* __restrict__ out, int rows, int D_) {
  int i = blockIdx.x * blockDim.x + threadIdx.x;
  if (i >= rows * D_) return;
  int r = i / D_, d = i - r * D_;
  out[i] = (__bf16)emb[idx[r] * D_ + d];
}

// ---------------------------------------------------------------------------
// Generic batched bf16 WMMA GEMM:  D[M,N] = act(scale * A[M,K] @ Bt[N,K]^T + bias) (+C)
//
// Register-blocked: each wave computes a 16x64 strip (4 accumulators), reusing
// one A fragment across 4 B fragments -> 4 back-to-back v_wmma per A load.
// Block = 8 waves laid out 4 (M) x 2 (N) => 64x128 tile per block.
// All GEMM shapes in this model have M,N multiples of 64, so no intra-wave
// edge handling is required (whole-wave skip only).
// Batch index z = bo*B2 + bi with independent outer/inner strides (batch/head).
// ---------------------------------------------------------------------------
__global__ void __launch_bounds__(256) k_gemm_bf16(
    const __bf16* __restrict__ A, long long sA1, long long sA2, int lda,
    const __bf16* __restrict__ Bt, long long sB1, long long sB2, int ldb,
    const float* __restrict__ bias,
    float* __restrict__ C, long long sC1, long long sC2, int ldc,
    __bf16* __restrict__ Cb, long long sD1, long long sD2, int ldd,
    int M, int N, int K, int B2, float scale, int act, int acc_c) {
  int z = blockIdx.z;
  int bo = z / B2, bi = z - bo * B2;
  int wave = threadIdx.x >> 5;
  int lane = threadIdx.x & 31;
  int row0 = (blockIdx.y * 4 + (wave >> 1)) * 16;   // 16-row strip
  int col0 = (blockIdx.x * 2 + (wave & 1)) * 64;    // 64-col strip (4 tiles)
  if (row0 >= M || col0 >= N) return;
  int half = lane >> 4;  // which 16-lane half
  int mr = lane & 15;    // row (A) / column (B) handled by this lane

  // A fragment element i: k = (i>>3)*16 + half*8 + (i&7)  -> two contiguous 16B loads
  const __bf16* Ap = A + bo * sA1 + bi * sA2 + (long long)(row0 + mr) * lda + half * 8;
  // B fragment element i: k = (i>>3)*16 + half*8 + (i&7)  -> two contiguous 16B loads
  const __bf16* Bp = Bt + bo * sB1 + bi * sB2 + (long long)(col0 + mr) * ldb + half * 8;

  v8f zero = {0.f, 0.f, 0.f, 0.f, 0.f, 0.f, 0.f, 0.f};
  v8f acc[4];
#pragma unroll
  for (int t = 0; t < 4; ++t) acc[t] = zero;

  for (int k0 = 0; k0 < K; k0 += 32) {
    __builtin_prefetch(Ap + k0 + 128, 0, 0);  // global_prefetch: next K-blocks of A
    v8bf alo = *(const v8bf*)(Ap + k0);
    v8bf ahi = *(const v8bf*)(Ap + k0 + 16);
    v16bf afrag;
#pragma unroll
    for (int i = 0; i < 8; ++i) {
      afrag[i] = alo[i];
      afrag[8 + i] = ahi[i];
    }
#pragma unroll
    for (int t = 0; t < 4; ++t) {
      v8bf blo = *(const v8bf*)(Bp + (long long)t * 16 * ldb + k0);
      v8bf bhi = *(const v8bf*)(Bp + (long long)t * 16 * ldb + k0 + 16);
      v16bf bfrag;
#pragma unroll
      for (int i = 0; i < 8; ++i) {
        bfrag[i] = blo[i];
        bfrag[8 + i] = bhi[i];
      }
      acc[t] = __builtin_amdgcn_wmma_f32_16x16x32_bf16(false, afrag, false, bfrag,
                                                       (short)0, acc[t], false, false);
    }
  }

#pragma unroll
  for (int t = 0; t < 4; ++t) {
    int n = col0 + t * 16 + mr;
    float bv = bias ? bias[n] : 0.0f;
#pragma unroll
    for (int r = 0; r < 8; ++r) {
      int m = row0 + half * 8 + r;  // C/D layout: lanes 0-15 -> M=r, lanes 16-31 -> M=8+r
      float v = acc[t][r] * scale + bv;
      if (act == 1) v = tanhf(v);
      else if (act == 2) v = fmaxf(v, 0.0f);
      else if (act == 3) v = 0.5f * v * (1.0f + erff(v * 0.70710678118f));
      if (C) {
        float* cp = C + bo * sC1 + bi * sC2 + (long long)m * ldc + n;
        if (acc_c) v += *cp;
        *cp = v;
      }
      if (Cb) Cb[bo * sD1 + bi * sD2 + (long long)m * ldd + n] = (__bf16)v;
    }
  }
}

// ---------------------------------------------------------------------------
// alpha[row] = proj_bf16[row,:] . fc1_w + fc1_b
// ---------------------------------------------------------------------------
__global__ void __launch_bounds__(256) k_alpha(const __bf16* __restrict__ proj,
                                               const float* __restrict__ w,
                                               const float* __restrict__ b,
                                               float* __restrict__ alpha, int K) {
  int row = blockIdx.x;
  __shared__ float red[256];
  float s = 0.f;
  for (int k = threadIdx.x; k < K; k += 256)
    s += (float)proj[(long long)row * K + k] * w[k];
  red[threadIdx.x] = s;
  __syncthreads();
  for (int off = 128; off > 0; off >>= 1) {
    if (threadIdx.x < off) red[threadIdx.x] += red[threadIdx.x + off];
    __syncthreads();
  }
  if (threadIdx.x == 0) alpha[row] = red[0] + b[0];
}

// ---------------------------------------------------------------------------
// Segment softmax-reduce (region pooling)
// ---------------------------------------------------------------------------
__device__ __forceinline__ void atomicMaxF(float* addr, float val) {
  int* ai = (int*)addr;
  int cur = *ai;
  while (__int_as_float(cur) < val) {
    int old = atomicCAS(ai, cur, __float_as_int(val));
    if (old == cur) break;
    cur = old;
  }
}

__global__ void k_seg_max(const float* __restrict__ alpha, const int* __restrict__ rid,
                          const bool* __restrict__ msk, float* __restrict__ amax,
                          int L_, int R_, int n) {
  int i = blockIdx.x * blockDim.x + threadIdx.x;
  if (i >= n || !msk[i]) return;
  atomicMaxF(&amax[(i / L_) * R_ + rid[i]], alpha[i]);
}

__global__ void k_seg_den(const float* __restrict__ alpha, const int* __restrict__ rid,
                          const bool* __restrict__ msk, const float* __restrict__ amax,
                          float* __restrict__ den, float* __restrict__ cnt,
                          int L_, int R_, int n) {
  int i = blockIdx.x * blockDim.x + threadIdx.x;
  if (i >= n || !msk[i]) return;
  int seg = (i / L_) * R_ + rid[i];
  atomicAdd(&den[seg], __expf(alpha[i] - amax[seg]));
  atomicAdd(&cnt[seg], 1.0f);
}

__global__ void k_seg_feats(const float* __restrict__ lf, const float* __restrict__ alpha,
                            const int* __restrict__ rid, const bool* __restrict__ msk,
                            const float* __restrict__ amax, const float* __restrict__ den,
                            float* __restrict__ rf, int L_, int R_, int D_, long long n) {
  long long i = (long long)blockIdx.x * blockDim.x + threadIdx.x;
  if (i >= n) return;
  int line = (int)(i / D_);
  int d = (int)(i - (long long)line * D_);
  if (!msk[line]) return;
  int seg = (line / L_) * R_ + rid[line];
  float w = __expf(alpha[line] - amax[seg]) / fmaxf(den[seg], 1e-20f);
  atomicAdd(&rf[(long long)seg * D_ + d], lf[i] * w);
}

// ---------------------------------------------------------------------------
// LayerNorm over last dim; fp32 in, bf16 out.  One row per block.
// ---------------------------------------------------------------------------
__global__ void __launch_bounds__(256) k_layernorm_bf16out(
    const float* __restrict__ x, const float* __restrict__ g, const float* __restrict__ b,
    __bf16* __restrict__ out, int D_) {
  int row = blockIdx.x;
  const float* xr = x + (long long)row * D_;
  __shared__ float red[256];
  __shared__ float s_mean, s_rstd;
  float s = 0.f;
  for (int d = threadIdx.x; d < D_; d += 256) s += xr[d];
  red[threadIdx.x] = s;
  __syncthreads();
  for (int off = 128; off > 0; off >>= 1) {
    if (threadIdx.x < off) red[threadIdx.x] += red[threadIdx.x + off];
    __syncthreads();
  }
  if (threadIdx.x == 0) s_mean = red[0] / D_;
  __syncthreads();
  float mean = s_mean;
  float v = 0.f;
  for (int d = threadIdx.x; d < D_; d += 256) {
    float t = xr[d] - mean;
    v += t * t;
  }
  red[threadIdx.x] = v;
  __syncthreads();
  for (int off = 128; off > 0; off >>= 1) {
    if (threadIdx.x < off) red[threadIdx.x] += red[threadIdx.x + off];
    __syncthreads();
  }
  if (threadIdx.x == 0) s_rstd = rsqrtf(red[0] / D_ + 1e-5f);
  __syncthreads();
  float rstd = s_rstd;
  __bf16* orow = out + (long long)row * D_;
  for (int d = threadIdx.x; d < D_; d += 256)
    orow[d] = (__bf16)((xr[d] - mean) * rstd * g[d] + b[d]);
}

// ---------------------------------------------------------------------------
// Masked row softmax over attention scores; fp32 in, bf16 out.  blockDim = R.
// ---------------------------------------------------------------------------
__global__ void __launch_bounds__(128) k_softmax_mask(const float* __restrict__ s,
                                                      const float* __restrict__ cnt,
                                                      __bf16* __restrict__ attn,
                                                      int H_, int R_) {
  int row = blockIdx.x;  // (b*H + h)*R + i
  int b = row / (H_ * R_);
  int j = threadIdx.x;
  const float* sp = s + (long long)row * R_;
  float m = (cnt[b * R_ + j] > 0.f) ? sp[j] : -1e9f;
  __shared__ float red[128];
  red[j] = m;
  __syncthreads();
  for (int off = 64; off > 0; off >>= 1) {
    if (j < off) red[j] = fmaxf(red[j], red[j + off]);
    __syncthreads();
  }
  float mx = red[0];
  __syncthreads();
  float e = __expf(m - mx);
  red[j] = e;
  __syncthreads();
  for (int off = 64; off > 0; off >>= 1) {
    if (j < off) red[j] += red[j + off];
    __syncthreads();
  }
  attn[(long long)row * R_ + j] = (__bf16)(e / red[0]);
}

// ---------------------------------------------------------------------------
// Final pair mask + diagonal mask
// ---------------------------------------------------------------------------
__global__ void k_mask_scores(const float* __restrict__ raw, const float* __restrict__ cnt,
                              float* __restrict__ out, int R_, int n) {
  int i = blockIdx.x * blockDim.x + threadIdx.x;
  if (i >= n) return;
  int b = i / (R_ * R_);
  int rem = i - b * R_ * R_;
  int r = rem / R_, c = rem - r * R_;
  float v = raw[i];
  if (!(cnt[b * R_ + r] > 0.f && cnt[b * R_ + c] > 0.f) || r == c) v = -1e9f;
  out[i] = v;
}

// ---------------------------------------------------------------------------
// Host orchestration
// ---------------------------------------------------------------------------
extern "C" void kernel_launch(void* const* d_in, const int* in_sizes, int n_in,
                              void* d_out, int out_size, void* d_ws, size_t ws_size,
                              hipStream_t stream) {
  (void)in_sizes; (void)n_in; (void)out_size; (void)ws_size;
  const int TB = 16, TL = 1024, TR = 128, TD = 768, TH = 12, THD = 64;
  const int TATT = 1024, TPRJ = 2048, TFFN = 2048, NL = 3;
  const int BL = TB * TL;  // 16384 lines
  const int BR = TB * TR;  // 2048 regions

  const float* lf    = (const float*)d_in[0];
  const int*   rid   = (const int*)d_in[1];
  const int*   roles = (const int*)d_in[2];
  const bool*  lmask = (const bool*)d_in[3];
  const float* fc2w  = (const float*)d_in[4];
  const float* fc2b  = (const float*)d_in[5];
  const float* fc1w  = (const float*)d_in[6];
  const float* fc1b  = (const float*)d_in[7];
  const float* temb  = (const float*)d_in[8];
  const float* tfcw  = (const float*)d_in[9];
  const float* tfcb  = (const float*)d_in[10];
  const float* tlng  = (const float*)d_in[11];
  const float* tlnb  = (const float*)d_in[12];
  const float* cw    = (const float*)d_in[13];
  const float* cb    = (const float*)d_in[14];
  const float* eWq   = (const float*)d_in[15];
  const float* ebq   = (const float*)d_in[16];
  const float* eWk   = (const float*)d_in[17];
  const float* ebk   = (const float*)d_in[18];
  const float* eWv   = (const float*)d_in[19];
  const float* ebv   = (const float*)d_in[20];
  const float* eWo   = (const float*)d_in[21];
  const float* ebo   = (const float*)d_in[22];
  const float* ln1g  = (const float*)d_in[23];
  const float* ln1b  = (const float*)d_in[24];
  const float* ln2g  = (const float*)d_in[25];
  const float* ln2b  = (const float*)d_in[26];
  const float* eW1   = (const float*)d_in[27];
  const float* eb1   = (const float*)d_in[28];
  const float* eW2   = (const float*)d_in[29];
  const float* eb2   = (const float*)d_in[30];
  const float* hw    = (const float*)d_in[31];
  const float* hb    = (const float*)d_in[32];
  const float* dwp   = (const float*)d_in[33];
  const float* db    = (const float*)d_in[34];
  const float* biaf  = (const float*)d_in[35];
  float* out = (float*)d_out;

  size_t cur = 0;
  auto alloc = [&](size_t bytes) -> void* {
    cur = (cur + 255) & ~(size_t)255;
    void* p = (char*)d_ws + cur;
    cur += bytes;
    return p;
  };

  __bf16* lf_bf   = (__bf16*)alloc((size_t)BL * TD * 2);
  __bf16* fc2t    = (__bf16*)alloc((size_t)TATT * TD * 2);
  __bf16* proj_bf = (__bf16*)alloc((size_t)BL * TATT * 2);
  float*  alpha   = (float*)alloc((size_t)BL * 4);
  float*  amax    = (float*)alloc((size_t)BR * 4);
  float*  den     = (float*)alloc((size_t)BR * 4);
  float*  cnt     = (float*)alloc((size_t)BR * 4);
  float*  rf      = (float*)alloc((size_t)BR * TD * 4);
  __bf16* rf_bf   = (__bf16*)alloc((size_t)BR * TD * 2);
  __bf16* g_bf    = (__bf16*)alloc((size_t)BR * TD * 2);
  __bf16* tfct    = (__bf16*)alloc((size_t)TD * TD * 2);
  float*  temb_f  = (float*)alloc((size_t)BR * TD * 4);
  __bf16* temb_bf = (__bf16*)alloc((size_t)BR * TD * 2);
  __bf16* cwt     = (__bf16*)alloc((size_t)TD * 2 * TD * 2);
  float*  x       = (float*)alloc((size_t)BR * TD * 4);
  __bf16* h_bf    = (__bf16*)alloc((size_t)BR * TD * 2);
  __bf16* wqt     = (__bf16*)alloc((size_t)TD * TD * 2);
  __bf16* wkt     = (__bf16*)alloc((size_t)TD * TD * 2);
  __bf16* wvt     = (__bf16*)alloc((size_t)TD * TD * 2);
  __bf16* wot     = (__bf16*)alloc((size_t)TD * TD * 2);
  __bf16* w1t     = (__bf16*)alloc((size_t)TFFN * TD * 2);
  __bf16* w2t     = (__bf16*)alloc((size_t)TD * TFFN * 2);
  __bf16* q_bf    = (__bf16*)alloc((size_t)BR * TD * 2);
  __bf16* k_bf    = (__bf16*)alloc((size_t)BR * TD * 2);
  __bf16* v_bf    = (__bf16*)alloc((size_t)BR * TD * 2);
  __bf16* vT      = (__bf16*)alloc((size_t)TB * TD * TR * 2);
  float*  sc      = (float*)alloc((size_t)TB * TH * TR * TR * 4);
  __bf16* att_bf  = (__bf16*)alloc((size_t)TB * TH * TR * TR * 2);
  __bf16* o_bf    = (__bf16*)alloc((size_t)BR * TD * 2);
  __bf16* f_bf    = (__bf16*)alloc((size_t)BR * TFFN * 2);
  __bf16* x_bf    = (__bf16*)alloc((size_t)BR * TD * 2);
  __bf16* hwt     = (__bf16*)alloc((size_t)TPRJ * TD * 2);
  __bf16* dwt     = (__bf16*)alloc((size_t)TPRJ * TD * 2);
  __bf16* bft     = (__bf16*)alloc((size_t)TPRJ * TPRJ * 2);
  __bf16* hq_bf   = (__bf16*)alloc((size_t)BR * TPRJ * 2);
  __bf16* hk_bf   = (__bf16*)alloc((size_t)BR * TPRJ * 2);
  __bf16* t_bf    = (__bf16*)alloc((size_t)BR * TPRJ * 2);
  float*  raw     = (float*)alloc((size_t)TB * TR * TR * 4);

  const int thr = 256;
  auto gemm = [&](const __bf16* A, long long sA1, long long sA2, int lda,
                  const __bf16* Bt, long long sB1, long long sB2, int ldb,
                  const float* bias, float* C, long long sC1, long long sC2, int ldc,
                  __bf16* Cb, long long sD1, long long sD2, int ldd,
                  int M, int N, int K, int B1, int B2, float scale, int act, int acc) {
    dim3 g(cdiv(N, 128), cdiv(M, 64), B1 * B2);
    k_gemm_bf16<<<g, dim3(256), 0, stream>>>(A, sA1, sA2, lda, Bt, sB1, sB2, ldb, bias,
                                             C, sC1, sC2, ldc, Cb, sD1, sD2, ldd,
                                             M, N, K, B2, scale, act, acc);
  };

  // ---- Region attention fusion ----
  k_f32_to_bf16<<<cdiv((long long)BL * TD, thr), thr, 0, stream>>>(lf, lf_bf, BL * TD);
  k_transpose_f32_to_bf16<<<cdiv((long long)TD * TATT, thr), thr, 0, stream>>>(fc2w, fc2t, TD, TATT);
  gemm(lf_bf, 0, 0, TD, fc2t, 0, 0, TD, fc2b, nullptr, 0, 0, 0, proj_bf, 0, 0, TATT,
       BL, TATT, TD, 1, 1, 1.f, 1 /*tanh*/, 0);
  k_alpha<<<BL, 256, 0, stream>>>(proj_bf, fc1w, fc1b, alpha, TATT);

  k_fill<<<cdiv(BR, thr), thr, 0, stream>>>(amax, -1e30f, BR);
  k_fill<<<cdiv(BR, thr), thr, 0, stream>>>(den, 0.f, BR);
  k_fill<<<cdiv(BR, thr), thr, 0, stream>>>(cnt, 0.f, BR);
  k_fill<<<cdiv((long long)BR * TD, thr), thr, 0, stream>>>(rf, 0.f, BR * TD);
  k_seg_max<<<cdiv(BL, thr), thr, 0, stream>>>(alpha, rid, lmask, amax, TL, TR, BL);
  k_seg_den<<<cdiv(BL, thr), thr, 0, stream>>>(alpha, rid, lmask, amax, den, cnt, TL, TR, BL);
  k_seg_feats<<<cdiv((long long)BL * TD, thr), thr, 0, stream>>>(lf, alpha, rid, lmask, amax, den,
                                                                 rf, TL, TR, TD, (long long)BL * TD);
  k_f32_to_bf16<<<cdiv((long long)BR * TD, thr), thr, 0, stream>>>(rf, rf_bf, BR * TD);

  // ---- Region type embedding: LN(ReLU(emb @ W + b)) ----
  k_gather_emb_bf16<<<cdiv((long long)BR * TD, thr), thr, 0, stream>>>(roles, temb, g_bf, BR, TD);
  k_transpose_f32_to_bf16<<<cdiv((long long)TD * TD, thr), thr, 0, stream>>>(tfcw, tfct, TD, TD);
  gemm(g_bf, 0, 0, TD, tfct, 0, 0, TD, tfcb, temb_f, 0, 0, TD, nullptr, 0, 0, 0,
       BR, TD, TD, 1, 1, 1.f, 2 /*relu*/, 0);
  k_layernorm_bf16out<<<BR, 256, 0, stream>>>(temb_f, tlng, tlnb, temb_bf, TD);

  // ---- Combine: x = concat(rf, temb) @ combine_w + b (two K=768 GEMMs) ----
  k_transpose_f32_to_bf16<<<cdiv((long long)2 * TD * TD, thr), thr, 0, stream>>>(cw, cwt, 2 * TD, TD);
  gemm(rf_bf, 0, 0, TD, cwt, 0, 0, 2 * TD, cb, x, 0, 0, TD, nullptr, 0, 0, 0,
       BR, TD, TD, 1, 1, 1.f, 0, 0);
  gemm(temb_bf, 0, 0, TD, cwt + TD, 0, 0, 2 * TD, nullptr, x, 0, 0, TD, nullptr, 0, 0, 0,
       BR, TD, TD, 1, 1, 1.f, 0, 1 /*accumulate*/);

  // ---- 3-layer pre-norm transformer ----
  for (int l = 0; l < NL; ++l) {
    const long long wsz = (long long)TD * TD;
    k_transpose_f32_to_bf16<<<cdiv(wsz, thr), thr, 0, stream>>>(eWq + l * wsz, wqt, TD, TD);
    k_transpose_f32_to_bf16<<<cdiv(wsz, thr), thr, 0, stream>>>(eWk + l * wsz, wkt, TD, TD);
    k_transpose_f32_to_bf16<<<cdiv(wsz, thr), thr, 0, stream>>>(eWv + l * wsz, wvt, TD, TD);
    k_transpose_f32_to_bf16<<<cdiv(wsz, thr), thr, 0, stream>>>(eWo + l * wsz, wot, TD, TD);
    k_transpose_f32_to_bf16<<<cdiv((long long)TD * TFFN, thr), thr, 0, stream>>>(
        eW1 + (long long)l * TD * TFFN, w1t, TD, TFFN);
    k_transpose_f32_to_bf16<<<cdiv((long long)TFFN * TD, thr), thr, 0, stream>>>(
        eW2 + (long long)l * TFFN * TD, w2t, TFFN, TD);

    k_layernorm_bf16out<<<BR, 256, 0, stream>>>(x, ln1g + l * TD, ln1b + l * TD, h_bf, TD);
    gemm(h_bf, 0, 0, TD, wqt, 0, 0, TD, ebq + l * TD, nullptr, 0, 0, 0, q_bf, 0, 0, TD,
         BR, TD, TD, 1, 1, 1.f, 0, 0);
    gemm(h_bf, 0, 0, TD, wkt, 0, 0, TD, ebk + l * TD, nullptr, 0, 0, 0, k_bf, 0, 0, TD,
         BR, TD, TD, 1, 1, 1.f, 0, 0);
    gemm(h_bf, 0, 0, TD, wvt, 0, 0, TD, ebv + l * TD, nullptr, 0, 0, 0, v_bf, 0, 0, TD,
         BR, TD, TD, 1, 1, 1.f, 0, 0);
    k_transpose_v_heads<<<cdiv((long long)BR * TD, thr), thr, 0, stream>>>(v_bf, vT, TB, TR, TD);

    // scores[b,h] = (q_bh @ k_bh^T) / sqrt(hd)
    gemm(q_bf, (long long)TR * TD, THD, TD, k_bf, (long long)TR * TD, THD, TD, nullptr,
         sc, (long long)TH * TR * TR, (long long)TR * TR, TR, nullptr, 0, 0, 0,
         TR, TR, THD, TB, TH, 0.125f, 0, 0);
    k_softmax_mask<<<TB * TH * TR, TR, 0, stream>>>(sc, cnt, att_bf, TH, TR);
    // o[b,h] = attn_bh @ v_bh   (B-fragments read from per-head-transposed V)
    gemm(att_bf, (long long)TH * TR * TR, (long long)TR * TR, TR,
         vT, (long long)TD * TR, (long long)THD * TR, TR, nullptr,
         nullptr, 0, 0, 0, o_bf, (long long)TR * TD, THD, TD,
         TR, THD, TR, TB, TH, 1.f, 0, 0);
    // x += o @ Wo + bo
    gemm(o_bf, 0, 0, TD, wot, 0, 0, TD, ebo + l * TD, x, 0, 0, TD, nullptr, 0, 0, 0,
         BR, TD, TD, 1, 1, 1.f, 0, 1);

    k_layernorm_bf16out<<<BR, 256, 0, stream>>>(x, ln2g + l * TD, ln2b + l * TD, h_bf, TD);
    gemm(h_bf, 0, 0, TD, w1t, 0, 0, TD, eb1 + l * TFFN, nullptr, 0, 0, 0, f_bf, 0, 0, TFFN,
         BR, TFFN, TD, 1, 1, 1.f, 3 /*gelu*/, 0);
    gemm(f_bf, 0, 0, TFFN, w2t, 0, 0, TFFN, eb2 + l * TD, x, 0, 0, TD, nullptr, 0, 0, 0,
         BR, TD, TFFN, 1, 1, 1.f, 0, 1);
  }

  // ---- Biaffine order head ----
  k_f32_to_bf16<<<cdiv((long long)BR * TD, thr), thr, 0, stream>>>(x, x_bf, BR * TD);
  k_transpose_f32_to_bf16<<<cdiv((long long)TD * TPRJ, thr), thr, 0, stream>>>(hw, hwt, TD, TPRJ);
  k_transpose_f32_to_bf16<<<cdiv((long long)TD * TPRJ, thr), thr, 0, stream>>>(dwp, dwt, TD, TPRJ);
  gemm(x_bf, 0, 0, TD, hwt, 0, 0, TD, hb, nullptr, 0, 0, 0, hq_bf, 0, 0, TPRJ,
       BR, TPRJ, TD, 1, 1, 1.f, 0, 0);
  gemm(x_bf, 0, 0, TD, dwt, 0, 0, TD, db, nullptr, 0, 0, 0, hk_bf, 0, 0, TPRJ,
       BR, TPRJ, TD, 1, 1, 1.f, 0, 0);
  k_transpose_f32_to_bf16<<<cdiv((long long)TPRJ * TPRJ, thr), thr, 0, stream>>>(biaf, bft, TPRJ, TPRJ);
  // t = hq @ biaffine   (big GEMM: 2048x2048x2048)
  gemm(hq_bf, 0, 0, TPRJ, bft, 0, 0, TPRJ, nullptr, nullptr, 0, 0, 0, t_bf, 0, 0, TPRJ,
       BR, TPRJ, TPRJ, 1, 1, 1.f, 0, 0);
  // raw[b] = (t_b @ hk_b^T) * PROJ^-0.5
  gemm(t_bf, (long long)TR * TPRJ, 0, TPRJ, hk_bf, (long long)TR * TPRJ, 0, TPRJ, nullptr,
       raw, (long long)TR * TR, 0, TR, nullptr, 0, 0, 0,
       TR, TR, TPRJ, TB, 1, 0.022097086912079608f, 0, 0);
  k_mask_scores<<<cdiv((long long)TB * TR * TR, thr), thr, 0, stream>>>(raw, cnt, out, TR, TB * TR * TR);
}
